// JaccardIndex_88905823027748
// MI455X (gfx1250) — compile-verified, matrix operand order checked by reading, not run on hardware
//
#include <hip/hip_runtime.h>
#include <math.h>

typedef float __attribute__((ext_vector_type(4))) f4;
typedef int   __attribute__((ext_vector_type(8))) v8i;

#define B_   8
#define C_   21
#define HW_  (512 * 512)
#define SPLITS 8
#define EPB  (HW_ / SPLITS)            // 32768 elements per block
#define EPI  (256 * 8 * 4)             // 8192 elements per block-iteration
#define ITERS (EPB / EPI)              // 4

// ---- zero the 63 per-class counters (workspace is poisoned by harness) ----
__global__ __launch_bounds__(64) void jaccard_zero(unsigned* __restrict__ ws) {
    int i = threadIdx.x;
    if (i < 3 * C_) ws[i] = 0u;
}

// ---- main streaming count kernel: WMMA-IU8 byte-count reduction ----
__global__ __launch_bounds__(256) void jaccard_count(const float* __restrict__ preds,
                                                     const float* __restrict__ target,
                                                     unsigned* __restrict__ ws) {
    const int chunk = blockIdx.x;            // chunk = b*C + c  (contiguous HW slab)
    const int c     = chunk % C_;
    const size_t base0 = (size_t)chunk * HW_ + (size_t)blockIdx.y * EPB;

    v8i accI = {};  // inter accumulator (16x16 i32)
    v8i accP = {};  // pred-count accumulator
    v8i accT = {};  // target-count accumulator
    v8i bOnes;
#pragma unroll
    for (int j = 0; j < 8; ++j) bOnes[j] = 0x01010101;  // B = all-ones 64x16 u8

    for (int it = 0; it < ITERS; ++it) {
        const size_t ibase = base0 + (size_t)it * EPI;
        v8i aP, aT, aI;
#pragma unroll
        for (int s = 0; s < 8; ++s) {
            // consecutive lanes -> consecutive float4s: fully coalesced 128b loads
            const size_t idx = ibase + ((size_t)(s * 256 + threadIdx.x)) * 4;
            f4 p  = __builtin_nontemporal_load((const f4*)(preds  + idx));
            f4 tg = __builtin_nontemporal_load((const f4*)(target + idx));
            unsigned pk = (unsigned)(p.x  >= 0.5f)
                        | ((unsigned)(p.y  >= 0.5f) << 8)
                        | ((unsigned)(p.z  >= 0.5f) << 16)
                        | ((unsigned)(p.w  >= 0.5f) << 24);
            unsigned tk = (unsigned)(tg.x == 1.0f)
                        | ((unsigned)(tg.y == 1.0f) << 8)
                        | ((unsigned)(tg.z == 1.0f) << 16)
                        | ((unsigned)(tg.w == 1.0f) << 24);
            aP[s] = (int)pk;
            aT[s] = (int)tk;
            aI[s] = (int)(pk & tk);   // bytes are 0/1, AND == logical-and per byte
        }
        // D = A x ones + C : every D element is a row-sum of A; total(D) = 16*count(A)
        accI = __builtin_amdgcn_wmma_i32_16x16x64_iu8(false, aI, false, bOnes, accI, false, false);
        accP = __builtin_amdgcn_wmma_i32_16x16x64_iu8(false, aP, false, bOnes, accP, false, false);
        accT = __builtin_amdgcn_wmma_i32_16x16x64_iu8(false, aT, false, bOnes, accT, false, false);
    }

    // per-lane horizontal sum of the 8 accumulator VGPRs
    int sI = 0, sP = 0, sT = 0;
#pragma unroll
    for (int j = 0; j < 8; ++j) { sI += accI[j]; sP += accP[j]; sT += accT[j]; }

    // wave32 butterfly reduction
#pragma unroll
    for (int off = 16; off > 0; off >>= 1) {
        sI += __shfl_xor(sI, off, 32);
        sP += __shfl_xor(sP, off, 32);
        sT += __shfl_xor(sT, off, 32);
    }

    __shared__ int partI[8], partP[8], partT[8];
    const int wave = threadIdx.x >> 5;
    const int lane = threadIdx.x & 31;
    if (lane == 0) { partI[wave] = sI; partP[wave] = sP; partT[wave] = sT; }
    __syncthreads();

    if (threadIdx.x == 0) {
        int tI = 0, tP = 0, tT = 0;
#pragma unroll
        for (int w = 0; w < 8; ++w) { tI += partI[w]; tP += partP[w]; tT += partT[w]; }
        // each wave total is exactly 16x the true count (16 identical D columns)
        atomicAdd(&ws[c * 3 + 0], (unsigned)(tI >> 4));
        atomicAdd(&ws[c * 3 + 1], (unsigned)(tP >> 4));
        atomicAdd(&ws[c * 3 + 2], (unsigned)(tT >> 4));
    }
}

// ---- finalize: 21 IoUs ----
__global__ __launch_bounds__(32) void jaccard_final(const unsigned* __restrict__ ws,
                                                    float* __restrict__ out) {
    int c = threadIdx.x;
    if (c < C_) {
        float inter = (float)ws[c * 3 + 0];
        float ps    = (float)ws[c * 3 + 1];
        float ts    = (float)ws[c * 3 + 2];
        float un    = ps + ts - inter;
        out[c] = (un == 0.0f) ? __builtin_nanf("") : (inter / fmaxf(un, 1.0f));
    }
}

extern "C" void kernel_launch(void* const* d_in, const int* in_sizes, int n_in,
                              void* d_out, int out_size, void* d_ws, size_t ws_size,
                              hipStream_t stream) {
    const float* preds  = (const float*)d_in[0];
    const float* target = (const float*)d_in[1];
    unsigned* ws  = (unsigned*)d_ws;
    float*    out = (float*)d_out;

    jaccard_zero<<<1, 64, 0, stream>>>(ws);
    jaccard_count<<<dim3(B_ * C_, SPLITS), 256, 0, stream>>>(preds, target, ws);
    jaccard_final<<<1, 32, 0, stream>>>(ws, out);
}
